// TimeSeriesModel_31250182045999
// MI455X (gfx1250) — matrix operand order repaired and checked by reading, not compiled, not run on hardware
//
#include <hip/hip_runtime.h>
#include <hip/hip_bf16.h>

// ---------------------------------------------------------------------------
// Informer ProbSparse attention on gfx1250 (MI455X), bf16 WMMA + flash softmax
// + double-buffered LDS with GLOBAL_LOAD_ASYNC_TO_LDS staging.
// B=4, L=2048, D=512, H=8, hd=64, n_top=7
// ---------------------------------------------------------------------------

#define B_ 4
#define L_ 2048
#define D_ 512
#define H_ 8
#define HD_ 64
#define NTOP 7
#define BH_ (B_ * H_)
#define M_ROWS (B_ * L_)   // 8192
#define SCALE_Q 0.125f     // 1/sqrt(64)

typedef __attribute__((ext_vector_type(16))) __bf16 bf16x16;
typedef __attribute__((ext_vector_type(8)))  __bf16 bf16x8;
typedef __attribute__((ext_vector_type(8)))  float  f32x8;

typedef __attribute__((address_space(3))) const void* lds_cvp;

static __device__ __forceinline__ f32x8 wmma_bf16(bf16x16 a, bf16x16 b, f32x8 c) {
  // D = A(16x32 bf16) * B(32x16 bf16) + C(16x16 f32)
  return __builtin_amdgcn_wmma_f32_16x16x32_bf16(false, a, false, b, (short)0, c,
                                                 false, false);
}

// Raw LDS byte offset of a __shared__ object (addrspace(3) ptrtoint).
static __device__ __forceinline__ unsigned lds_off(const void* p) {
  return (unsigned)(unsigned long long)(lds_cvp)p;
}

// 16-byte async copy global -> LDS (ASYNCcnt-tracked, no VGPR data).
static __device__ __forceinline__ void async_cp16(unsigned lds, const void* g) {
  asm volatile("global_load_async_to_lds_b128 %0, %1, off"
               :: "v"(lds), "v"((unsigned long long)g) : "memory");
}
static __device__ __forceinline__ void async_wait0() {
  asm volatile("s_wait_asynccnt 0x0" ::: "memory");
}

// A-operand gather for 16-bit 16x32 A tile (row-major LDS row pointer).
// Lane layout: lane l<16 holds row l, K in {0..15};
// lane l>=16 holds row l-16, K in {16..31} (matches contiguous B loads).
static __device__ __forceinline__ bf16x16 ldA(const __bf16* rowp, int hf) {
  bf16x8 lo = *(const bf16x8*)(rowp + hf * 16);
  bf16x8 hi = *(const bf16x8*)(rowp + hf * 16 + 8);
  return __builtin_shufflevector(lo, hi, 0, 1, 2, 3, 4, 5, 6, 7,
                                 8, 9, 10, 11, 12, 13, 14, 15);
}

static __device__ __forceinline__ bf16x8 cvt8(float4 a, float4 b) {
  bf16x8 v = {(__bf16)a.x, (__bf16)a.y, (__bf16)a.z, (__bf16)a.w,
              (__bf16)b.x, (__bf16)b.y, (__bf16)b.z, (__bf16)b.w};
  return v;
}

// ---------------------------------------------------------------------------
// 1. weight conversion: w_in -> wq/wk/wv bf16 [N=512][K=512]; w_out -> wo
// ---------------------------------------------------------------------------
__global__ __launch_bounds__(256) void k_cvt_w(const float* __restrict__ w_in,
                                               const float* __restrict__ w_out,
                                               __bf16* __restrict__ wq,
                                               __bf16* __restrict__ wk,
                                               __bf16* __restrict__ wv,
                                               __bf16* __restrict__ wo) {
  const int c = blockIdx.x * 256 + threadIdx.x;  // chunk of 8
  const int NK = D_ * D_;
  if (c * 8 < NK) {
    const float4* a;
    a = (const float4*)&w_in[c * 8];
    *(bf16x8*)&wq[c * 8] = cvt8(a[0], a[1]);
    a = (const float4*)&w_in[NK + c * 8];
    *(bf16x8*)&wk[c * 8] = cvt8(a[0], a[1]);
    a = (const float4*)&w_in[2 * NK + c * 8];
    *(bf16x8*)&wv[c * 8] = cvt8(a[0], a[1]);
    a = (const float4*)&w_out[c * 8];
    *(bf16x8*)&wo[c * 8] = cvt8(a[0], a[1]);
  }
}

// ---------------------------------------------------------------------------
// 2. raw q/k -> head-major bf16 [B,H,L,hd]  (8 elements per thread)
// ---------------------------------------------------------------------------
__global__ __launch_bounds__(256) void k_cvt_raw(const float* __restrict__ q,
                                                 const float* __restrict__ k,
                                                 __bf16* __restrict__ Qraw,
                                                 __bf16* __restrict__ Kraw) {
  const int c = blockIdx.x * 256 + threadIdx.x;  // chunk of 8 over 2^22 elems
  const int i = c * 8;
  const int j = i & (HD_ - 1);
  const int l = (i >> 6) & (L_ - 1);
  const int h = (i >> 17) & (H_ - 1);
  const int b = i >> 20;
  const size_t src = ((size_t)(b * L_ + l)) * D_ + h * HD_ + j;
  const float4* aq = (const float4*)&q[src];
  const float4* ak = (const float4*)&k[src];
  *(bf16x8*)&Qraw[i] = cvt8(aq[0], aq[1]);
  *(bf16x8*)&Kraw[i] = cvt8(ak[0], ak[1]);
}

// ---------------------------------------------------------------------------
// 3. projection GEMM: Y[m,n] = (sum_k X[m,k]*W[n,k] + bias[n]) * scale,
//    stored head-major bf16 [B,H,L,hd].  grid=(M/64, N/32), block=128.
//    Double-buffered LDS; W tile staged with async-to-LDS, X tile via regs
//    (needs fp32->bf16 conversion).
// ---------------------------------------------------------------------------
__global__ __launch_bounds__(128) void k_proj(const float* __restrict__ X,
                                              const __bf16* __restrict__ W,
                                              const float* __restrict__ bias,
                                              __bf16* __restrict__ Y,
                                              float scale) {
  __shared__ __align__(32) __bf16 At[2][64][32];
  __shared__ __align__(32) __bf16 Bt[2][32][32];
  const int t = threadIdx.x, wid = t >> 5, ln = t & 31;
  const int hf = ln >> 4, r = ln & 15;
  const int m0 = blockIdx.x * 64, n0 = blockIdx.y * 32;
  const int ar = t >> 1, ac = (t & 1) * 16;   // A: row, col-base (16 floats)
  const int br = t >> 2, bc = (t & 3) * 8;    // B: row, col-base (8 bf16)
  f32x8 acc0 = {}, acc1 = {};
  // prologue: stage K-block 0 into buffer 0
  {
    const float4* xp = (const float4*)&X[(size_t)(m0 + ar) * D_ + ac];
    async_cp16(lds_off(&Bt[0][br][bc]), &W[(size_t)(n0 + br) * D_ + bc]);
    *(bf16x8*)&At[0][ar][ac] = cvt8(xp[0], xp[1]);
    *(bf16x8*)&At[0][ar][ac + 8] = cvt8(xp[2], xp[3]);
    async_wait0();
    __syncthreads();
  }
  for (int ki = 0; ki < D_ / 32; ++ki) {
    const int cur = ki & 1, nxt = cur ^ 1;
    const int k0n = (ki + 1) * 32;
    const bool have_next = k0n < D_;
    float4 x0, x1, x2, x3;
    if (have_next) {
      async_cp16(lds_off(&Bt[nxt][br][bc]), &W[(size_t)(n0 + br) * D_ + k0n + bc]);
      const float4* xp = (const float4*)&X[(size_t)(m0 + ar) * D_ + k0n + ac];
      x0 = xp[0]; x1 = xp[1]; x2 = xp[2]; x3 = xp[3];
    }
    // compute current buffer
    bf16x16 a = ldA(&At[cur][wid * 16 + r][0], hf);
    bf16x16 b0 = *(const bf16x16*)&Bt[cur][r][hf * 16];
    bf16x16 b1 = *(const bf16x16*)&Bt[cur][16 + r][hf * 16];
    acc0 = wmma_bf16(a, b0, acc0);
    acc1 = wmma_bf16(a, b1, acc1);
    if (have_next) {
      *(bf16x8*)&At[nxt][ar][ac] = cvt8(x0, x1);
      *(bf16x8*)&At[nxt][ar][ac + 8] = cvt8(x2, x3);
    }
    async_wait0();
    __syncthreads();
  }
#pragma unroll
  for (int g = 0; g < 8; ++g) {
    const int m = m0 + wid * 16 + g + 8 * hf;
    const int b = m >> 11, l = m & (L_ - 1);
#pragma unroll
    for (int half_n = 0; half_n < 2; ++half_n) {
      const int n = n0 + half_n * 16 + r;
      const float y = ((half_n ? acc1[g] : acc0[g]) + bias[n]) * scale;
      const int h = n >> 6, j = n & (HD_ - 1);
      Y[(((size_t)(b * H_ + h)) * L_ + l) * HD_ + j] = (__bf16)y;
    }
  }
}

// ---------------------------------------------------------------------------
// 4. M-score streaming pass: M[l] = max_s(q_raw.k_raw) - mean_s(...)
//    grid=(L/64, B*H), block=128.  K tiles double-buffered via async-to-LDS.
// ---------------------------------------------------------------------------
__global__ __launch_bounds__(128) void k_mscore(const __bf16* __restrict__ Qraw,
                                                const __bf16* __restrict__ Kraw,
                                                float* __restrict__ Mout) {
  const int bh = blockIdx.y;
  const __bf16* Q = Qraw + (size_t)bh * L_ * HD_;
  const __bf16* K = Kraw + (size_t)bh * L_ * HD_;
  __shared__ __align__(32) __bf16 Qt[64][64];
  __shared__ __align__(32) __bf16 Kt[2][32][64];
  const int t = threadIdx.x, wid = t >> 5, ln = t & 31;
  const int hf = ln >> 4, r = ln & 15;
  const int q0 = blockIdx.x * 64;
#pragma unroll
  for (int i = 0; i < 4; ++i) {  // Qt: 512 chunks of 8, pure copy -> async
    const int ch = t + i * 128, rr = ch >> 3, cc = (ch & 7) * 8;
    async_cp16(lds_off(&Qt[rr][cc]), &Q[(size_t)(q0 + rr) * HD_ + cc]);
  }
#pragma unroll
  for (int i = 0; i < 2; ++i) {  // Kt block 0
    const int ch = t + i * 128, rr = ch >> 3, cc = (ch & 7) * 8;
    async_cp16(lds_off(&Kt[0][rr][cc]), &K[(size_t)rr * HD_ + cc]);
  }
  async_wait0();
  __syncthreads();
  // Q operands are loop-invariant: hoist
  bf16x16 a0 = ldA(&Qt[wid * 16 + r][0], hf);
  bf16x16 a1 = ldA(&Qt[wid * 16 + r][32], hf);
  float rmax[8], rsum[8];
#pragma unroll
  for (int g = 0; g < 8; ++g) { rmax[g] = -1e30f; rsum[g] = 0.f; }
  for (int blk = 0; blk < L_ / 32; ++blk) {
    const int cur = blk & 1, nxt = cur ^ 1;
    if (blk + 1 < L_ / 32) {
      const int s0n = (blk + 1) * 32;
#pragma unroll
      for (int i = 0; i < 2; ++i) {
        const int ch = t + i * 128, rr = ch >> 3, cc = (ch & 7) * 8;
        async_cp16(lds_off(&Kt[nxt][rr][cc]), &K[(size_t)(s0n + rr) * HD_ + cc]);
      }
    }
    f32x8 acc0 = {}, acc1 = {};
    {
      bf16x16 b00 = *(const bf16x16*)&Kt[cur][r][hf * 16];
      bf16x16 b10 = *(const bf16x16*)&Kt[cur][16 + r][hf * 16];
      bf16x16 b01 = *(const bf16x16*)&Kt[cur][r][32 + hf * 16];
      bf16x16 b11 = *(const bf16x16*)&Kt[cur][16 + r][32 + hf * 16];
      acc0 = wmma_bf16(a0, b00, acc0);
      acc0 = wmma_bf16(a1, b01, acc0);
      acc1 = wmma_bf16(a0, b10, acc1);
      acc1 = wmma_bf16(a1, b11, acc1);
    }
#pragma unroll
    for (int g = 0; g < 8; ++g) {
      float bm = fmaxf(acc0[g], acc1[g]);
      float bs = acc0[g] + acc1[g];
#pragma unroll
      for (int msk = 1; msk < 16; msk <<= 1) {
        bm = fmaxf(bm, __shfl_xor(bm, msk, 32));
        bs += __shfl_xor(bs, msk, 32);
      }
      rmax[g] = fmaxf(rmax[g], bm);
      rsum[g] += bs;
    }
    async_wait0();
    __syncthreads();
  }
  if (r == 0) {
#pragma unroll
    for (int g = 0; g < 8; ++g) {
      const int q = q0 + wid * 16 + g + 8 * hf;
      Mout[(size_t)bh * L_ + q] = rmax[g] - rsum[g] * (1.0f / (float)L_);
    }
  }
}

// ---------------------------------------------------------------------------
// 5. top-7 selection per (b,h): sel[l] = 1 if l among top-7 of M, else 0
// ---------------------------------------------------------------------------
__global__ __launch_bounds__(256) void k_topk(const float* __restrict__ Mbuf,
                                              float* __restrict__ sel) {
  const int bh = blockIdx.x;
  __shared__ float sM[L_];
  __shared__ float rv[256];
  __shared__ int ri[256];
  __shared__ int top[NTOP];
  const int t = threadIdx.x;
#pragma unroll
  for (int i = 0; i < 8; ++i) sM[t * 8 + i] = Mbuf[(size_t)bh * L_ + t * 8 + i];
  __syncthreads();
  for (int it = 0; it < NTOP; ++it) {
    float bv = -1e38f;
    int bi = 0;
#pragma unroll
    for (int i = 0; i < 8; ++i) {
      const int idx = t * 8 + i;
      const float v = sM[idx];
      if (v > bv) { bv = v; bi = idx; }
    }
    rv[t] = bv; ri[t] = bi;
    __syncthreads();
    for (int s = 128; s > 0; s >>= 1) {
      if (t < s) {
        if (rv[t + s] > rv[t] || (rv[t + s] == rv[t] && ri[t + s] < ri[t])) {
          rv[t] = rv[t + s]; ri[t] = ri[t + s];
        }
      }
      __syncthreads();
    }
    if (t == 0) { top[it] = ri[0]; sM[ri[0]] = -1e38f; }
    __syncthreads();
  }
#pragma unroll
  for (int i = 0; i < 8; ++i) {
    const int idx = t * 8 + i;
    float s = 0.f;
    for (int j = 0; j < NTOP; ++j)
      if (top[j] == idx) s = 1.f;
    sel[(size_t)bh * L_ + idx] = s;
  }
}

// ---------------------------------------------------------------------------
// 6. flash attention with augmented (projected + sel-masked raw) scores.
//    grid=(L/64, B*H), block=128.  Qp is pre-scaled by 1/sqrt(hd).
//    K tiles double-buffered via async-to-LDS; V transposed via registers.
// ---------------------------------------------------------------------------
__global__ __launch_bounds__(128) void k_attn(const __bf16* __restrict__ Qp,
                                              const __bf16* __restrict__ Kp,
                                              const __bf16* __restrict__ Vp,
                                              const __bf16* __restrict__ Qraw,
                                              const __bf16* __restrict__ Kraw,
                                              const float* __restrict__ sel,
                                              __bf16* __restrict__ ctx) {
  const int bh = blockIdx.y;
  const int b = bh >> 3, h = bh & (H_ - 1);
  const size_t base = (size_t)bh * L_ * HD_;
  __shared__ __align__(32) __bf16 Qt[64][64];        // projected (scaled)
  __shared__ __align__(32) __bf16 Qr[64][64];        // raw * sel(row)
  __shared__ __align__(32) __bf16 Kt[2][32][64];
  __shared__ __align__(32) __bf16 Kr[2][32][64];
  __shared__ __align__(32) __bf16 Vt[2][64][32];     // transposed [feat][key]
  __shared__ __align__(32) __bf16 Pt[4][16][32];     // per-wave P tile
  const int t = threadIdx.x, wid = t >> 5, ln = t & 31;
  const int hf = ln >> 4, r = ln & 15;
  const int q0 = blockIdx.x * 64;
  const bf16x8 zero8 = {};

  // stage K/Kraw (async) and V (register transpose) for a key block
  auto stage_k = [&](int buf, int s0) {
#pragma unroll
    for (int i = 0; i < 2; ++i) {
      const int ch = t + i * 128, rr = ch >> 3, cc = (ch & 7) * 8;
      const size_t go = base + (size_t)(s0 + rr) * HD_ + cc;
      async_cp16(lds_off(&Kt[buf][rr][cc]), &Kp[go]);
      async_cp16(lds_off(&Kr[buf][rr][cc]), &Kraw[go]);
    }
  };
  auto stage_v = [&](int buf, int s0) {
#pragma unroll
    for (int i = 0; i < 2; ++i) {
      const int ch = t + i * 128, rr = ch >> 3, cc = (ch & 7) * 8;
      bf16x8 v8 = *(const bf16x8*)&Vp[base + (size_t)(s0 + rr) * HD_ + cc];
#pragma unroll
      for (int j = 0; j < 8; ++j) Vt[buf][cc + j][rr] = v8[j];
    }
  };

  // prologue: Qt async copy, Qr sel-masked register copy, K/V block 0
#pragma unroll
  for (int i = 0; i < 4; ++i) {
    const int ch = t + i * 128, rr = ch >> 3, cc = (ch & 7) * 8;
    async_cp16(lds_off(&Qt[rr][cc]), &Qp[base + (size_t)(q0 + rr) * HD_ + cc]);
    bf16x8 qr8 = *(const bf16x8*)&Qraw[base + (size_t)(q0 + rr) * HD_ + cc];
    const float s = sel[(size_t)bh * L_ + q0 + rr];
    *(bf16x8*)&Qr[rr][cc] = (s != 0.f) ? qr8 : zero8;  // sel is exactly 0/1
  }
  stage_k(0, 0);
  stage_v(0, 0);
  async_wait0();
  __syncthreads();
  // Q operands are loop-invariant: hoist
  bf16x16 aP0 = ldA(&Qt[wid * 16 + r][0], hf);
  bf16x16 aP1 = ldA(&Qt[wid * 16 + r][32], hf);
  bf16x16 aR0 = ldA(&Qr[wid * 16 + r][0], hf);
  bf16x16 aR1 = ldA(&Qr[wid * 16 + r][32], hf);

  float mrow[8], dsum[8];
  f32x8 o0 = {}, o1 = {}, o2 = {}, o3 = {};
#pragma unroll
  for (int g = 0; g < 8; ++g) { mrow[g] = -1e30f; dsum[g] = 0.f; }

  for (int blk = 0; blk < L_ / 32; ++blk) {
    const int cur = blk & 1, nxt = cur ^ 1;
    if (blk + 1 < L_ / 32) {
      stage_k(nxt, (blk + 1) * 32);
      stage_v(nxt, (blk + 1) * 32);
    }
    // S(16 rows x 32 keys) = Qp.Kp^T (scaled) + Qraw_masked.Kraw^T
    f32x8 acc0 = {}, acc1 = {};
    {
      bf16x16 b0p0 = *(const bf16x16*)&Kt[cur][r][hf * 16];
      bf16x16 b0p1 = *(const bf16x16*)&Kt[cur][r][32 + hf * 16];
      bf16x16 b1p0 = *(const bf16x16*)&Kt[cur][16 + r][hf * 16];
      bf16x16 b1p1 = *(const bf16x16*)&Kt[cur][16 + r][32 + hf * 16];
      bf16x16 b0r0 = *(const bf16x16*)&Kr[cur][r][hf * 16];
      bf16x16 b0r1 = *(const bf16x16*)&Kr[cur][r][32 + hf * 16];
      bf16x16 b1r0 = *(const bf16x16*)&Kr[cur][16 + r][hf * 16];
      bf16x16 b1r1 = *(const bf16x16*)&Kr[cur][16 + r][32 + hf * 16];
      acc0 = wmma_bf16(aP0, b0p0, acc0);
      acc0 = wmma_bf16(aP1, b0p1, acc0);
      acc0 = wmma_bf16(aR0, b0r0, acc0);
      acc0 = wmma_bf16(aR1, b0r1, acc0);
      acc1 = wmma_bf16(aP0, b1p0, acc1);
      acc1 = wmma_bf16(aP1, b1p1, acc1);
      acc1 = wmma_bf16(aR0, b1r0, acc1);
      acc1 = wmma_bf16(aR1, b1r1, acc1);
    }
    // online softmax update over the 32-key block
#pragma unroll
    for (int g = 0; g < 8; ++g) {
      float bm = fmaxf(acc0[g], acc1[g]);
#pragma unroll
      for (int msk = 1; msk < 16; msk <<= 1)
        bm = fmaxf(bm, __shfl_xor(bm, msk, 32));
      const float mn = fmaxf(mrow[g], bm);
      const float p0 = __expf(acc0[g] - mn);
      const float p1 = __expf(acc1[g] - mn);
      float ps = p0 + p1;
#pragma unroll
      for (int msk = 1; msk < 16; msk <<= 1) ps += __shfl_xor(ps, msk, 32);
      const float al = __expf(mrow[g] - mn);
      dsum[g] = dsum[g] * al + ps;
      mrow[g] = mn;
      o0[g] *= al; o1[g] *= al; o2[g] *= al; o3[g] *= al;
      Pt[wid][g + 8 * hf][r] = (__bf16)p0;
      Pt[wid][g + 8 * hf][16 + r] = (__bf16)p1;
    }
    // O += P(16x32) @ V(32x64)
    bf16x16 aPV = ldA(&Pt[wid][r][0], hf);
    bf16x16 bv0 = *(const bf16x16*)&Vt[cur][0 + r][hf * 16];
    bf16x16 bv1 = *(const bf16x16*)&Vt[cur][16 + r][hf * 16];
    bf16x16 bv2 = *(const bf16x16*)&Vt[cur][32 + r][hf * 16];
    bf16x16 bv3 = *(const bf16x16*)&Vt[cur][48 + r][hf * 16];
    o0 = wmma_bf16(aPV, bv0, o0);
    o1 = wmma_bf16(aPV, bv1, o1);
    o2 = wmma_bf16(aPV, bv2, o2);
    o3 = wmma_bf16(aPV, bv3, o3);
    async_wait0();
    __syncthreads();
  }
  // normalize and scatter to [B, L, D] (head-interleaved) bf16
#pragma unroll
  for (int g = 0; g < 8; ++g) {
    const int q = q0 + wid * 16 + g + 8 * hf;
    const float inv = 1.0f / dsum[g];
    const size_t ro = ((size_t)(b * L_ + q)) * D_ + h * HD_;
    ctx[ro + 0 + r]  = (__bf16)(o0[g] * inv);
    ctx[ro + 16 + r] = (__bf16)(o1[g] * inv);
    ctx[ro + 32 + r] = (__bf16)(o2[g] * inv);
    ctx[ro + 48 + r] = (__bf16)(o3[g] * inv);
  }
}

// ---------------------------------------------------------------------------
// 7. output projection: out[m,n] = sum_k ctx[m,k]*wo[n,k] + b_out[n] (fp32 out)
//    grid=(M/64, N/32), block=128.  Fully async-staged double buffer.
// ---------------------------------------------------------------------------
__global__ __launch_bounds__(128) void k_outproj(const __bf16* __restrict__ X,
                                                 const __bf16* __restrict__ W,
                                                 const float* __restrict__ bias,
                                                 float* __restrict__ Y) {
  __shared__ __align__(32) __bf16 At[2][64][32];
  __shared__ __align__(32) __bf16 Bt[2][32][32];
  const int t = threadIdx.x, wid = t >> 5, ln = t & 31;
  const int hf = ln >> 4, r = ln & 15;
  const int m0 = blockIdx.x * 64, n0 = blockIdx.y * 32;
  const int br = t >> 2, bc = (t & 3) * 8;
  auto stage = [&](int buf, int k0) {
#pragma unroll
    for (int i = 0; i < 2; ++i) {  // At: 256 chunks of 8
      const int ch = t + i * 128, rr = ch >> 2, cc = (ch & 3) * 8;
      async_cp16(lds_off(&At[buf][rr][cc]), &X[(size_t)(m0 + rr) * D_ + k0 + cc]);
    }
    async_cp16(lds_off(&Bt[buf][br][bc]), &W[(size_t)(n0 + br) * D_ + k0 + bc]);
  };
  f32x8 acc0 = {}, acc1 = {};
  stage(0, 0);
  async_wait0();
  __syncthreads();
  for (int ki = 0; ki < D_ / 32; ++ki) {
    const int cur = ki & 1, nxt = cur ^ 1;
    const int k0n = (ki + 1) * 32;
    if (k0n < D_) stage(nxt, k0n);
    bf16x16 a = ldA(&At[cur][wid * 16 + r][0], hf);
    bf16x16 b0 = *(const bf16x16*)&Bt[cur][r][hf * 16];
    bf16x16 b1 = *(const bf16x16*)&Bt[cur][16 + r][hf * 16];
    acc0 = wmma_bf16(a, b0, acc0);
    acc1 = wmma_bf16(a, b1, acc1);
    async_wait0();
    __syncthreads();
  }
#pragma unroll
  for (int g = 0; g < 8; ++g) {
    const int m = m0 + wid * 16 + g + 8 * hf;
    Y[(size_t)m * D_ + n0 + r] = acc0[g] + bias[n0 + r];
    Y[(size_t)m * D_ + n0 + 16 + r] = acc1[g] + bias[n0 + 16 + r];
  }
}

// ---------------------------------------------------------------------------
// host launcher
// ---------------------------------------------------------------------------
extern "C" void kernel_launch(void* const* d_in, const int* in_sizes, int n_in,
                              void* d_out, int out_size, void* d_ws, size_t ws_size,
                              hipStream_t stream) {
  const float* query = (const float*)d_in[0];
  const float* key_i = (const float*)d_in[1];
  const float* value = (const float*)d_in[2];
  const float* w_in  = (const float*)d_in[3];
  const float* b_in  = (const float*)d_in[4];
  const float* w_out = (const float*)d_in[5];
  const float* b_out = (const float*)d_in[6];
  float* out = (float*)d_out;

  // workspace carve-up (~51 MB total)
  size_t off = 0;
  auto alloc = [&](size_t bytes) {
    void* p = (char*)d_ws + off;
    off += (bytes + 255) & ~(size_t)255;
    return p;
  };
  const size_t NKW = (size_t)D_ * D_ * sizeof(__bf16);            // 512 KB
  const size_t NHM = (size_t)B_ * H_ * L_ * HD_ * sizeof(__bf16); // 8 MB
  const size_t NBL = (size_t)B_ * H_ * L_ * sizeof(float);        // 256 KB
  __bf16* wq   = (__bf16*)alloc(NKW);
  __bf16* wk   = (__bf16*)alloc(NKW);
  __bf16* wv   = (__bf16*)alloc(NKW);
  __bf16* wo   = (__bf16*)alloc(NKW);
  __bf16* Qraw = (__bf16*)alloc(NHM);
  __bf16* Kraw = (__bf16*)alloc(NHM);
  __bf16* Qp   = (__bf16*)alloc(NHM);
  __bf16* Kp   = (__bf16*)alloc(NHM);
  __bf16* Vp   = (__bf16*)alloc(NHM);
  __bf16* ctx  = (__bf16*)alloc(NHM);
  float*  Mbuf = (float*)alloc(NBL);
  float*  sel  = (float*)alloc(NBL);
  (void)ws_size; (void)in_sizes; (void)n_in; (void)out_size;

  // 1-2. conversions (vectorized: 8 elems/thread)
  k_cvt_w<<<(D_ * D_ / 8 + 255) / 256, 256, 0, stream>>>(w_in, w_out, wq, wk, wv, wo);
  k_cvt_raw<<<(B_ * H_ * L_ * HD_ / 8) / 256, 256, 0, stream>>>(query, key_i, Qraw, Kraw);

  // 3. input projections (Q pre-scaled by 1/sqrt(hd))
  dim3 ggemm(M_ROWS / 64, D_ / 32, 1);
  k_proj<<<ggemm, 128, 0, stream>>>(query, wq, b_in + 0 * D_, Qp, SCALE_Q);
  k_proj<<<ggemm, 128, 0, stream>>>(key_i, wk, b_in + 1 * D_, Kp, 1.0f);
  k_proj<<<ggemm, 128, 0, stream>>>(value, wv, b_in + 2 * D_, Vp, 1.0f);

  // 4-5. sparsity measure + top-7 mask
  dim3 gatt(L_ / 64, BH_, 1);
  k_mscore<<<gatt, 128, 0, stream>>>(Qraw, Kraw, Mbuf);
  k_topk<<<BH_, 256, 0, stream>>>(Mbuf, sel);

  // 6. flash attention with augmented scores
  k_attn<<<gatt, 128, 0, stream>>>(Qp, Kp, Vp, Qraw, Kraw, sel, ctx);

  // 7. output projection -> fp32 d_out
  k_outproj<<<ggemm, 128, 0, stream>>>(ctx, wo, b_out, out);
}
